// MixedOp_37099927503005
// MI455X (gfx1250) — matrix operand order repaired and checked by reading, not compiled, not run on hardware
//
#include <hip/hip_runtime.h>

// ---------------------------------------------------------------------------
// DARTS MixedOp forward for MI455X (gfx1250, wave32, WMMA).
// x: [64,128,32,32] f32; weights: [64,8]; out = mixed tensor + logits copy.
// Pointwise 1x1 convs are per-batch 128x128x1024 GEMMs -> v_wmma_f32_16x16x32_f16.
// ---------------------------------------------------------------------------

typedef __attribute__((ext_vector_type(16))) _Float16 v16h;
typedef __attribute__((ext_vector_type(8)))  _Float16 v8h;
typedef __attribute__((ext_vector_type(8)))  float    v8f;

#define BN_SCALE 0.99999500003749981f   // 1/sqrt(1 + 1e-5)

#define NB   64
#define NC   128
#define NH   32
#define NW   32
#define NHW  1024
#define NCHW 131072           // 2^17
#define NOPS 8

// ---------------------------------------------------------------------------
// Gate: top-k over 8 logits per batch, softmax over the selected set.
// Non-selected ops get an exact 0.0f gate (enables uniform block skips).
// Also copies logits into the second output (reference returns `weights`).
// ---------------------------------------------------------------------------
__global__ void gate_topk_kernel(const float* __restrict__ wts,
                                 const int* __restrict__ top_p,
                                 float* __restrict__ gate,
                                 float* __restrict__ w_out) {
  int b = blockIdx.x * blockDim.x + threadIdx.x;
  if (b >= NB) return;
  float lg[NOPS];
#pragma unroll
  for (int i = 0; i < NOPS; ++i) lg[i] = wts[b * NOPS + i];

  int top = top_p[0];
  if (top > NOPS) top = NOPS;
  bool sel[NOPS];
#pragma unroll
  for (int i = 0; i < NOPS; ++i) sel[i] = false;
  for (int t = 0; t < top; ++t) {          // repeated argmax, ties -> lowest index
    int best = -1;
    float bv = -__builtin_inff();
    for (int i = 0; i < NOPS; ++i)
      if (!sel[i] && lg[i] > bv) { bv = lg[i]; best = i; }
    if (best >= 0) sel[best] = true;
  }
  float mx = -__builtin_inff();
#pragma unroll
  for (int i = 0; i < NOPS; ++i) if (sel[i] && lg[i] > mx) mx = lg[i];
  float e[NOPS];
  float den = 0.f;
#pragma unroll
  for (int i = 0; i < NOPS; ++i) {
    e[i] = sel[i] ? __expf(lg[i] - mx) : 0.f;
    den += e[i];
  }
  float inv = 1.f / den;
#pragma unroll
  for (int i = 0; i < NOPS; ++i) {
    gate[b * NOPS + i] = e[i] * inv;
    w_out[b * NOPS + i] = lg[i];
  }
}

// ---------------------------------------------------------------------------
// Fused maxpool3x3 + avgpool3x3 (count_include_pad=False) + skip_connect.
// Initializes d_out:  out = g1*BN*max + g2*BN*avg + g3*x   (g0 'none' = 0).
// One block never straddles a batch boundary -> gate branch is uniform.
// ---------------------------------------------------------------------------
__global__ void pool_skip_kernel(const float* __restrict__ x,
                                 const float* __restrict__ gate,
                                 float* __restrict__ out) {
  unsigned idx = blockIdx.x * 256u + threadIdx.x;   // [0, NB*NCHW)
  unsigned b  = idx >> 17;
  unsigned hw = idx & (NHW - 1);
  int h = (int)(hw >> 5);
  int w = (int)(hw & 31);
  float g1 = gate[b * NOPS + 1];
  float g2 = gate[b * NOPS + 2];
  float g3 = gate[b * NOPS + 3];

  float v = 0.f;
  if (g1 != 0.f || g2 != 0.f) {
    const float* plane = x + (idx - hw);   // base of this (b,c) plane
    float mx = -__builtin_inff();
    float sum = 0.f;
    int cnt = 0;
#pragma unroll
    for (int dy = -1; dy <= 1; ++dy) {
      int hh = h + dy;
      if ((unsigned)hh < (unsigned)NH) {
#pragma unroll
        for (int dx = -1; dx <= 1; ++dx) {
          int ww = w + dx;
          if ((unsigned)ww < (unsigned)NW) {
            float t = plane[hh * NW + ww];
            mx = fmaxf(mx, t);
            sum += t;
            ++cnt;
          }
        }
      }
    }
    v = g1 * (BN_SCALE * mx) + g2 * (BN_SCALE * (sum / (float)cnt));
  }
  v += g3 * x[idx];
  out[idx] = v;
}

// ---------------------------------------------------------------------------
// Fused [ReLU -> depthwise KSxKS (dilation DIL) -> pointwise 1x1 GEMM -> BN].
// One workgroup = (batch b, 4-row spatial tile = 128 positions).
//   Phase 1: pw weights -> LDS f16  (A matrix, [co][ci], padded stride)
//            + global_prefetch of the input rows phase 2 will read
//   Phase 2: dw(relu(in)) -> LDS f16 (B matrix, [s][ci], padded stride)
//   Phase 3: 8 waves x (8 N-tiles x 4 K-steps) v_wmma_f32_16x16x32_f16.
// ACCUM=false: write y = BN*gemm to workspace (sep stage 1).
// ACCUM=true : d_out += gate*BN*gemm (sep stage 2 / dil conv).
// Uniform early-out when gate==0 (EXEC stays all-ones around WMMA).
// Epilogue: one 64-bit base per lane; all loads/stores use immediate offsets
// (nt*64B + r*4096B), eliminating per-element 64-bit address math.
// ---------------------------------------------------------------------------
template <int KS, int DIL, bool ACCUM>
__global__ __launch_bounds__(256)
void sepstage_kernel(const float* __restrict__ in,    // [B,C,H,W]
                     const float* __restrict__ dw,    // [C,1,KS,KS]
                     const float* __restrict__ pw,    // [C,C,1,1]
                     const float* __restrict__ gate,  // [B,8]
                     int op_idx,
                     float* __restrict__ out) {
  constexpr int LDW = NC + 8;                 // padded stride in halves (272B rows)
  constexpr int HALO = (KS / 2) * DIL;        // rows of halo above/below the tile
  __shared__ _Float16 lds_w[NC * LDW];        // A: [co][ci]
  __shared__ _Float16 lds_b[NC * LDW];        // B: [s][ci]
  __shared__ float    lds_dw[NC * KS * KS];

  const int b  = blockIdx.x >> 3;
  const int r0 = (blockIdx.x & 7) * 4;        // first of 4 rows in this tile
  const int tid = threadIdx.x;

  const float g = gate[b * NOPS + op_idx];
  if (g == 0.f) return;                       // uniform across the block

  const float* inb = in + (size_t)b * NCHW;

  // ---- Phase 1: stage weights into LDS; prefetch phase-2 input rows ----
  {
    const int rlo = (r0 - HALO) < 0 ? 0 : (r0 - HALO);
    const int rhi = (r0 + 3 + HALO) > (NH - 1) ? (NH - 1) : (r0 + 3 + HALO);
    const int nrows = rhi - rlo + 1;
    for (int i = tid; i < NC * nrows; i += 256) {
      int ci = i / nrows, rr = i - ci * nrows + rlo;
      __builtin_prefetch(inb + ci * NHW + rr * NW, 0, 1);  // global_prefetch_b8
    }
  }
  for (int i = tid; i < NC * NC; i += 256) {
    int co = i >> 7, ci = i & (NC - 1);
    lds_w[co * LDW + ci] = (_Float16)pw[i];
  }
  for (int i = tid; i < NC * KS * KS; i += 256) lds_dw[i] = dw[i];
  __syncthreads();

  // ---- Phase 2: depthwise conv (ReLU fused) into LDS ------------------
  for (int i = tid; i < NC * 128; i += 256) {
    int ci = i >> 7;
    int s  = i & 127;
    int h = r0 + (s >> 5);
    int w = s & 31;
    const float* xp = inb + ci * NHW;
    const float* kw = &lds_dw[ci * KS * KS];
    float acc = 0.f;
#pragma unroll
    for (int ky = 0; ky < KS; ++ky) {
      int hh = h + (ky - KS / 2) * DIL;
      if ((unsigned)hh < (unsigned)NH) {
#pragma unroll
        for (int kx = 0; kx < KS; ++kx) {
          int ww = w + (kx - KS / 2) * DIL;
          if ((unsigned)ww < (unsigned)NW) {
            float t = xp[hh * NW + ww];
            acc += (t > 0.f ? t : 0.f) * kw[ky * KS + kx];
          }
        }
      }
    }
    lds_b[s * LDW + ci] = (_Float16)acc;
  }
  __syncthreads();

  // ---- Phase 3: 128x128x128 GEMM via WMMA -----------------------------
  const int wave  = tid >> 5;
  const int lane  = tid & 31;
  const int lhalf = lane >> 4;       // 0: lanes 0-15, 1: lanes 16-31
  const int lmod  = lane & 15;
  const int m0    = wave * 16;       // this wave's M strip (out channels)

  // Preload the 4 A fragments (K = 0..127 in steps of 32).
  // 16-bit A 16x32 layout: lane<16 holds K k0..k0+7 then k0+16..k0+23;
  // lanes 16-31 are shifted by +8 in K.  Two ds_load_b128 per fragment.
  v16h afrag[4];
  const int arow = m0 + lmod;
#pragma unroll
  for (int kt = 0; kt < 4; ++kt) {
    const int k0 = kt * 32 + lhalf * 8;
    const v8h lo = *(const v8h*)&lds_w[arow * LDW + k0];
    const v8h hi = *(const v8h*)&lds_w[arow * LDW + k0 + 16];
#pragma unroll
    for (int j = 0; j < 8; ++j) { afrag[kt][j] = lo[j]; afrag[kt][j + 8] = hi[j]; }
  }

  // One 64-bit output base per lane; every access below is base + imm.
  // D layout: VGPR r -> M = m0 + r (+8 for lanes 16-31), N = lmod.
  // Tile nt, row r lives at   base + nt*16 + r*1024   floats.
  const int   co0   = m0 + lhalf * 8;
  float*      outp  = out + ((size_t)(b * NC + co0) * NHW) + r0 * NW + lmod;
  const float scale = ACCUM ? (g * BN_SCALE) : BN_SCALE;

#pragma unroll
  for (int nt = 0; nt < 8; ++nt) {
    const int n0 = nt * 16;
    const int brow = n0 + lmod;      // spatial column owned by this lane
    v8f acc = {};
#pragma unroll
    for (int kt = 0; kt < 4; ++kt) {
      // 16-bit B 32x16 layout (matches A): lane<16 -> K k0..k0+7, k0+16..k0+23; lanes 16-31 +8.
      const int k0 = kt * 32 + lhalf * 8;
      const v8h lo = *(const v8h*)&lds_b[brow * LDW + k0];
      const v8h hi = *(const v8h*)&lds_b[brow * LDW + k0 + 16];
      v16h bfrag;
#pragma unroll
      for (int j = 0; j < 8; ++j) { bfrag[j] = lo[j]; bfrag[j + 8] = hi[j]; }
      acc = __builtin_amdgcn_wmma_f32_16x16x32_f16(
          /*neg_a=*/false, afrag[kt], /*neg_b=*/false, bfrag,
          /*c_mod=*/(short)0, acc, /*reuse_a=*/false, /*reuse_b=*/false);
    }
    float* p = outp + nt * 16;
#pragma unroll
    for (int r = 0; r < 8; ++r) {
      if (ACCUM) p[r * NHW] = fmaf(scale, acc[r], p[r * NHW]);
      else       p[r * NHW] = scale * acc[r];
    }
  }
}

// ---------------------------------------------------------------------------
// Launch: gate -> pool/skip (writes out) -> sep3 (2 stages) -> sep5 (2 stages)
//         -> dil3 -> dil5  (accumulating, stream-ordered).
// ---------------------------------------------------------------------------
extern "C" void kernel_launch(void* const* d_in, const int* in_sizes, int n_in,
                              void* d_out, int out_size, void* d_ws, size_t ws_size,
                              hipStream_t stream) {
  (void)in_sizes; (void)n_in; (void)out_size; (void)ws_size;

  const float* x      = (const float*)d_in[0];
  const float* wts    = (const float*)d_in[1];
  const float* sc3_d1 = (const float*)d_in[2];
  const float* sc3_p1 = (const float*)d_in[3];
  const float* sc3_d2 = (const float*)d_in[4];
  const float* sc3_p2 = (const float*)d_in[5];
  const float* sc5_d1 = (const float*)d_in[6];
  const float* sc5_p1 = (const float*)d_in[7];
  const float* sc5_d2 = (const float*)d_in[8];
  const float* sc5_p2 = (const float*)d_in[9];
  const float* dc3_d  = (const float*)d_in[10];
  const float* dc3_p  = (const float*)d_in[11];
  const float* dc5_d  = (const float*)d_in[12];
  const float* dc5_p  = (const float*)d_in[13];
  const int*   top    = (const int*)d_in[14];

  float* out   = (float*)d_out;
  float* w_out = out + (size_t)NB * NCHW;         // second output: logits copy

  float* gate = (float*)d_ws;                     // NB*8 floats
  float* ybuf = (float*)((char*)d_ws + 4096);     // [B,C,H,W] f32 intermediate

  const int gemm_grid = NB * 8;                   // (batch, 4-row tile)

  gate_topk_kernel<<<1, 64, 0, stream>>>(wts, top, gate, w_out);
  pool_skip_kernel<<<(NB * NCHW) / 256, 256, 0, stream>>>(x, gate, out);

  // sep_conv_3x3 (op 4): two relu-dw-pw-BN stages
  sepstage_kernel<3, 1, false><<<gemm_grid, 256, 0, stream>>>(x,    sc3_d1, sc3_p1, gate, 4, ybuf);
  sepstage_kernel<3, 1, true ><<<gemm_grid, 256, 0, stream>>>(ybuf, sc3_d2, sc3_p2, gate, 4, out);
  // sep_conv_5x5 (op 5)
  sepstage_kernel<5, 1, false><<<gemm_grid, 256, 0, stream>>>(x,    sc5_d1, sc5_p1, gate, 5, ybuf);
  sepstage_kernel<5, 1, true ><<<gemm_grid, 256, 0, stream>>>(ybuf, sc5_d2, sc5_p2, gate, 5, out);
  // dil_conv_3x3 (op 6), dil_conv_5x5 (op 7): single stage each
  sepstage_kernel<3, 2, true ><<<gemm_grid, 256, 0, stream>>>(x, dc3_d, dc3_p, gate, 6, out);
  sepstage_kernel<5, 2, true ><<<gemm_grid, 256, 0, stream>>>(x, dc5_d, dc5_p, gate, 7, out);
}
